// BertSelfAttention_687194768194
// MI455X (gfx1250) — matrix operand order repaired and checked by reading, not compiled, not run on hardware
//
#include <hip/hip_runtime.h>
#include <hip/hip_bf16.h>

typedef __attribute__((ext_vector_type(16))) _Float16 v16h;
typedef __attribute__((ext_vector_type(8)))  float    v8f;

#define NB    2
#define SEQ   2048
#define DMOD  768
#define NH    12
#define DHD   64
#define MROWS (NB * SEQ)          // 4096

// ---------------- fragment helpers (wave32 WMMA, contiguous k = 16*g + s for A and B) ----------------

// A-matrix 16x32 f16: lane -> row m = lane&15, group g = lane>>4,
// VGPR pair p holds K = 16*g + 2*p  (+1); contiguous k order chosen here, matched by B (ISA table differs).
// Per lane this is one contiguous 32-byte run.
__device__ __forceinline__ v16h load_a16x32(const _Float16* tile, int ld) {
    const int lane = threadIdx.x & 31;
    const int m = lane & 15;
    const int g = lane >> 4;
    const _Float16* row = tile + (size_t)m * ld;
    v16h a;
#pragma unroll
    for (int p = 0; p < 8; ++p) {
        const int k0 = 16 * g + 2 * p;
        a[2 * p]     = row[k0];
        a[2 * p + 1] = row[k0 + 1];
    }
    return a;
}

// B-matrix 32x16 f16: lane holds column n = lane&15; halves h=0..15 are
// K = 16*(lane>>4) + h. With column-contiguous storage this is one 32-byte read.
__device__ __forceinline__ v16h load_b32x16(const _Float16* src, int ld) {
    const int lane = threadIdx.x & 31;
    const int n = lane & 15;
    const int g = lane >> 4;
    return *(const v16h*)(src + (size_t)n * ld + 16 * g);
}

__device__ __forceinline__ v8f wmma_f16(v16h a, v16h b, v8f c) {
    return __builtin_amdgcn_wmma_f32_16x16x32_f16(false, a, false, b, (short)0, c, false, false);
}

// ---------------- kernel 0: fp32 -> f16 conversion ----------------

__global__ __launch_bounds__(256) void cvt_kernel(const float* __restrict__ X,
                                                  const float* __restrict__ Wq,
                                                  const float* __restrict__ Wk,
                                                  const float* __restrict__ Wv,
                                                  _Float16* __restrict__ Xh,
                                                  _Float16* __restrict__ Wh) {
    const int NX = MROWS * DMOD;       // 3,145,728
    const int NW = DMOD * DMOD;        // 589,824
    const int NT = NX + 3 * NW;
    for (int i = blockIdx.x * blockDim.x + threadIdx.x; i < NT; i += gridDim.x * blockDim.x) {
        if (i < NX) {
            Xh[i] = (_Float16)X[i];
        } else {
            int j = i - NX;
            const float* W = (j < NW) ? Wq : ((j < 2 * NW) ? Wk : Wv);
            Wh[j] = (_Float16)W[j % NW];
        }
    }
}

// ---------------- kernel 1: fused QKV projection GEMM ----------------
// grid: (MROWS/128, DMOD/64, 3), block: 128 (4 waves).
// Wave w computes a 32x64 output slab (2 M-tiles x 4 N-tiles):
// per K step: 2 A-frag + 4 B-frag loads -> 8 WMMAs (2x arithmetic intensity).

__global__ __launch_bounds__(128) void qkv_gemm_kernel(const _Float16* __restrict__ Xh,
                                                       const _Float16* __restrict__ Wh,
                                                       const float* __restrict__ bq,
                                                       const float* __restrict__ bk,
                                                       const float* __restrict__ bv,
                                                       _Float16* __restrict__ Qh,
                                                       _Float16* __restrict__ Kh,
                                                       _Float16* __restrict__ Vt) {
    const int w      = threadIdx.x >> 5;
    const int mslab  = blockIdx.x * 4 + w;     // 0..127, each = 32 rows
    const int n64    = blockIdx.y;             // 0..11
    const int which  = blockIdx.z;             // 0=Q 1=K 2=V

    const _Float16* W    = Wh + (size_t)which * DMOD * DMOD;
    const float*    bias = (which == 0) ? bq : ((which == 1) ? bk : bv);

    v8f acc[2][4] = {{v8f{}, v8f{}, v8f{}, v8f{}}, {v8f{}, v8f{}, v8f{}, v8f{}}};
    const _Float16* Xt0 = Xh + (size_t)(mslab * 2 + 0) * 16 * DMOD;
    const _Float16* Xt1 = Xh + (size_t)(mslab * 2 + 1) * 16 * DMOD;

    for (int k = 0; k < DMOD; k += 32) {
        v16h a0 = load_a16x32(Xt0 + k, DMOD);
        v16h a1 = load_a16x32(Xt1 + k, DMOD);
#pragma unroll
        for (int t = 0; t < 4; ++t) {
            const _Float16* Wt = W + (size_t)(n64 * 64 + t * 16) * DMOD + k;
            v16h b = load_b32x16(Wt, DMOD);
            acc[0][t] = wmma_f16(a0, b, acc[0][t]);
            acc[1][t] = wmma_f16(a1, b, acc[1][t]);
        }
    }

    const int lane = threadIdx.x & 31;
    const int g    = lane >> 4;
    const int ln   = lane & 15;
    const float scale = (which == 0) ? 0.125f : 1.0f;  // 1/sqrt(DH) folded into Q

#pragma unroll
    for (int mm = 0; mm < 2; ++mm) {
#pragma unroll
        for (int t = 0; t < 4; ++t) {
            const int ncol = n64 * 64 + t * 16 + ln;   // output feature [0,768)
            const int h    = ncol / DHD;
            const int dh   = ncol % DHD;
            const float bv_ = bias[ncol];
#pragma unroll
            for (int r = 0; r < 8; ++r) {
                const int m = (mslab * 2 + mm) * 16 + r + 8 * g; // global row [0,4096)
                const int b = m / SEQ;
                const int s = m % SEQ;
                const float v = (acc[mm][t][r] + bv_) * scale;
                if (which == 2)
                    Vt[(((size_t)b * NH + h) * DHD + dh) * SEQ + s] = (_Float16)v;
                else if (which == 0)
                    Qh[(((size_t)b * NH + h) * SEQ + s) * DHD + dh] = (_Float16)v;
                else
                    Kh[(((size_t)b * NH + h) * SEQ + s) * DHD + dh] = (_Float16)v;
            }
        }
    }
}

// ---------------- kernel 2: flash attention per 16-query tile ----------------
// grid: (SEQ/16/4, NB*NH), block 128 (4 waves). Wave handles one 16-row Q tile.

__global__ __launch_bounds__(128) void attn_kernel(const _Float16* __restrict__ Qh,
                                                   const _Float16* __restrict__ Kh,
                                                   const _Float16* __restrict__ Vt,
                                                   const float* __restrict__ mask,
                                                   float* __restrict__ out) {
    __shared__ __align__(64) _Float16 Plds[4][16][32];  // per-wave P tile (f16)
    __shared__ float rowf[4][16];                        // per-wave row broadcast

    const int w    = threadIdx.x >> 5;
    const int lane = threadIdx.x & 31;
    const int g    = lane >> 4;
    const int ln   = lane & 15;

    const int qtile = blockIdx.x * 4 + w;   // 0..127
    const int bh    = blockIdx.y;           // 0..23
    const int b     = bh / NH;
    const int h     = bh % NH;

    const _Float16* Q = Qh + (size_t)bh * SEQ * DHD;
    const _Float16* K = Kh + (size_t)bh * SEQ * DHD;
    const _Float16* V = Vt + (size_t)bh * DHD * SEQ;
    const float*  msk = mask + (size_t)b * SEQ;

    // Q fragments (scaled by 1/8 already): 16 rows x 64 (two k-halves)
    const v16h qa0 = load_a16x32(Q + (size_t)qtile * 16 * DHD, DHD);
    const v16h qa1 = load_a16x32(Q + (size_t)qtile * 16 * DHD + 32, DHD);

    v8f ctx[4] = {v8f{}, v8f{}, v8f{}, v8f{}};   // ctx^T: 64(dh) x 16(query)
    float rm[8], rs[8];
#pragma unroll
    for (int r = 0; r < 8; ++r) { rm[r] = -3.0e38f; rs[r] = 0.0f; }

    for (int j = 0; j < SEQ; j += 32) {
        // ---- scores for 32 keys: two 16x16 C-frags ----
        v16h kb00 = load_b32x16(K + (size_t)j * DHD, DHD);
        v16h kb01 = load_b32x16(K + (size_t)j * DHD + 32, DHD);
        v16h kb10 = load_b32x16(K + (size_t)(j + 16) * DHD, DHD);
        v16h kb11 = load_b32x16(K + (size_t)(j + 16) * DHD + 32, DHD);
        v8f s0 = {}, s1 = {};
        s0 = wmma_f16(qa0, kb00, s0);
        s0 = wmma_f16(qa1, kb01, s0);
        s1 = wmma_f16(qa0, kb10, s1);
        s1 = wmma_f16(qa1, kb11, s1);

        const float m0 = msk[j + ln];
        const float m1 = msk[j + 16 + ln];

        // ---- online softmax: row m = r + 8g spans 16 lanes at VGPR r ----
        float alpha[8];
#pragma unroll
        for (int r = 0; r < 8; ++r) {
            float v0 = s0[r] + m0;
            float v1 = s1[r] + m1;
            float mx = fmaxf(v0, v1);
#pragma unroll
            for (int off = 8; off; off >>= 1) mx = fmaxf(mx, __shfl_xor(mx, off, 16));
            const float nm = fmaxf(rm[r], mx);
            const float a  = __expf(rm[r] - nm);
            const float p0 = __expf(v0 - nm);
            const float p1 = __expf(v1 - nm);
            float sum = p0 + p1;
#pragma unroll
            for (int off = 8; off; off >>= 1) sum += __shfl_xor(sum, off, 16);
            rs[r] = rs[r] * a + sum;
            rm[r] = nm;
            alpha[r] = a;
            s0[r] = p0;
            s1[r] = p1;
        }

        // ---- P (C-layout) -> LDS -> B-fragment of P^T; broadcast alpha ----
#pragma unroll
        for (int r = 0; r < 8; ++r) {
            const int m = r + 8 * g;
            Plds[w][m][ln]      = (_Float16)s0[r];
            Plds[w][m][16 + ln] = (_Float16)s1[r];
        }
        if (ln == 0) {
#pragma unroll
            for (int r = 0; r < 8; ++r) rowf[w][r + 8 * g] = alpha[r];
        }
        // per-wave LDS is in-order: loads below observe the stores above
        const float am = rowf[w][ln];          // rescale for my query column
#pragma unroll
        for (int t = 0; t < 4; ++t)
#pragma unroll
            for (int r = 0; r < 8; ++r) ctx[t][r] *= am;

        const v16h pb = *(const v16h*)&Plds[w][ln][16 * g];  // P^T B-frag

        // ---- ctx^T += V^T(64x32) x P^T(32x16) : 4 WMMAs ----
#pragma unroll
        for (int t = 0; t < 4; ++t) {
            v16h va = load_a16x32(V + (size_t)(t * 16) * SEQ + j, SEQ);
            ctx[t] = wmma_f16(va, pb, ctx[t]);
        }
    }

    // ---- finalize: divide by row sums, store [B,S,H*DH] fp32 ----
    if (ln == 0) {
#pragma unroll
        for (int r = 0; r < 8; ++r) rowf[w][r + 8 * g] = rs[r];
    }
    const float inv = 1.0f / rowf[w][ln];

    const int srow = qtile * 16 + ln;          // my query row
    float* o = out + ((size_t)b * SEQ + srow) * (NH * DHD) + h * DHD;
#pragma unroll
    for (int t = 0; t < 4; ++t)
#pragma unroll
        for (int r = 0; r < 8; ++r)
            o[t * 16 + r + 8 * g] = ctx[t][r] * inv;
}

// ---------------- launcher ----------------

extern "C" void kernel_launch(void* const* d_in, const int* in_sizes, int n_in,
                              void* d_out, int out_size, void* d_ws, size_t ws_size,
                              hipStream_t stream) {
    const float* X    = (const float*)d_in[0];
    const float* mask = (const float*)d_in[1];
    const float* Wq   = (const float*)d_in[2];
    const float* bq   = (const float*)d_in[3];
    const float* Wk   = (const float*)d_in[4];
    const float* bk   = (const float*)d_in[5];
    const float* Wv   = (const float*)d_in[6];
    const float* bv   = (const float*)d_in[7];
    float* out = (float*)d_out;

    // workspace layout (halves)
    _Float16* Xh = (_Float16*)d_ws;
    _Float16* Wh = Xh + (size_t)MROWS * DMOD;                 // 3x [768,768]
    _Float16* Qh = Wh + (size_t)3 * DMOD * DMOD;              // [B,H,S,DH]
    _Float16* Kh = Qh + (size_t)MROWS * DMOD;                 // [B,H,S,DH]
    _Float16* Vt = Kh + (size_t)MROWS * DMOD;                 // [B,H,DH,S]

    cvt_kernel<<<4800, 256, 0, stream>>>(X, Wq, Wk, Wv, Xh, Wh);

    qkv_gemm_kernel<<<dim3(MROWS / 128, DMOD / 64, 3), 128, 0, stream>>>(
        Xh, Wh, bq, bk, bv, Qh, Kh, Vt);

    attn_kernel<<<dim3(SEQ / 16 / 4, NB * NH), 128, 0, stream>>>(
        Qh, Kh, Vt, mask, out);
}
